// AtomTransformerBlock_85478439125476
// MI455X (gfx1250) — matrix operand order repaired and checked by reading, not compiled, not run on hardware
//
#include <hip/hip_runtime.h>
#include <hip/hip_bf16.h>
#include <stdint.h>

#define N_ATOMS 2048
#define C_ATOM  128
#define C_PAIR  16
#define N_HEADS 4
#define C_HEAD  32
#define KT      32                      // key tile
#define TILE_F  (16 * KT * C_PAIR)      // floats per plm LDS tile (8192 = 32KB)

typedef __attribute__((ext_vector_type(16))) __bf16 v16bf;
typedef __attribute__((ext_vector_type(8)))  float  v8f;
typedef __attribute__((ext_vector_type(4)))  unsigned int u32x4;
typedef __attribute__((ext_vector_type(4)))  int    i32x4;
typedef __attribute__((ext_vector_type(8)))  int    i32x8;

union BFrag { v16bf v; uint32_t u[8]; };

__device__ __forceinline__ uint16_t f2bf(float f) {
  uint32_t u = __float_as_uint(f);
  uint32_t r = (u + 0x7FFFu + ((u >> 16) & 1u)) >> 16;
  return (uint16_t)r;
}

#define F_BIAS    1
#define F_SIGMOID 2
#define F_RELU    4
#define F_RESID   8

#if __has_builtin(__builtin_amdgcn_tensor_load_to_lds) && \
    __has_builtin(__builtin_amdgcn_s_wait_tensorcnt)
#define HAVE_TDM 1
#else
#define HAVE_TDM 0
#endif

#if HAVE_TDM
// Issue one TDM descriptor: copy a 16-row x (KT*C_PAIR)-float 2D tile of plm
// (row stride = N_ATOMS*C_PAIR elements) into LDS at lds_dst. D# per ISA ch.8.
__device__ __forceinline__ void tdm_load_plm_tile(const float* gsrc,
                                                  void* lds_dst) {
  uint64_t ga = (uint64_t)(uintptr_t)gsrc;
  uint32_t la = (uint32_t)(uintptr_t)lds_dst;  // LDS aperture: low 32 bits
  u32x4 g0;
  g0[0] = 1u;                                   // count=1, user descriptor
  g0[1] = la;                                   // lds_addr (bytes)
  g0[2] = (uint32_t)ga;                         // global_addr[31:0]
  g0[3] = (uint32_t)((ga >> 32) & 0x01FFFFFFu)  // global_addr[56:32]
          | (2u << 30);                         // type=2 ("image")
  i32x8 g1;
  const uint32_t dim0 = (uint32_t)(N_ATOMS * C_PAIR);   // 32768 elements
  const uint32_t dim1 = (uint32_t)N_ATOMS;              // 2048 rows
  g1[0] = (int)(2u << 16);                              // data_size=4B, mask=0
  g1[1] = (int)((dim0 & 0xFFFFu) << 16);                // tensor_dim0[15:0]
  g1[2] = (int)(((dim0 >> 16) & 0xFFFFu) | ((dim1 & 0xFFFFu) << 16));
  g1[3] = (int)(((dim1 >> 16) & 0xFFFFu) | ((uint32_t)(KT * C_PAIR) << 16));
  g1[4] = 16;                                           // tile_dim1=16, dim2=0
  g1[5] = (int)dim0;                                    // tensor_dim0_stride lo
  g1[6] = 0;                                            // stride hi / d1 stride
  g1[7] = 0;
  i32x4 z4 = {0, 0, 0, 0};                              // groups 2/3 unused
  i32x8 z8 = {0, 0, 0, 0, 0, 0, 0, 0};
  __builtin_amdgcn_tensor_load_to_lds(g0, g1, z4, z4, z8, 0);
}
#endif

// ---------------------------------------------------------------------------
// fp32 -> bf16 conversion
// ---------------------------------------------------------------------------
__global__ void cvt_bf16_kernel(const float* __restrict__ s,
                                uint16_t* __restrict__ d, int n) {
  int i = blockIdx.x * 256 + threadIdx.x;
  if (i < n) d[i] = f2bf(s[i]);
}

// ---------------------------------------------------------------------------
// Row LayerNorm over C=128, output bf16
// ---------------------------------------------------------------------------
__global__ void __launch_bounds__(128)
layernorm128_kernel(const float* __restrict__ x, const float* __restrict__ w,
                    const float* __restrict__ b, uint16_t* __restrict__ out) {
  __shared__ float red[128];
  const int row = blockIdx.x;
  const int t = threadIdx.x;
  float v = x[(size_t)row * C_ATOM + t];
  red[t] = v;
  __syncthreads();
  for (int s = 64; s > 0; s >>= 1) {
    if (t < s) red[t] += red[t + s];
    __syncthreads();
  }
  float mu = red[0] * (1.0f / 128.0f);
  __syncthreads();
  float d = v - mu;
  red[t] = d * d;
  __syncthreads();
  for (int s = 64; s > 0; s >>= 1) {
    if (t < s) red[t] += red[t + s];
    __syncthreads();
  }
  float var = red[0] * (1.0f / 128.0f);
  float y = d * rsqrtf(var + 1e-5f) * w[t] + b[t];
  out[(size_t)row * C_ATOM + t] = f2bf(y);
}

// ---------------------------------------------------------------------------
// Generic bf16 WMMA GEMM:  out[m,n] = epi( sum_k A[m,k] * Bw[n,k] )
// ---------------------------------------------------------------------------
__global__ void __launch_bounds__(128)
gemm_bf16_wmma(const uint16_t* __restrict__ A, const uint16_t* __restrict__ Bw,
               const float* __restrict__ bias, const float* __restrict__ resid,
               float scale, int flags,
               float* __restrict__ outF, uint16_t* __restrict__ outB,
               uint16_t* __restrict__ outBT, int M, int N, int K) {
  const int wave = threadIdx.x >> 5;
  const int lane = threadIdx.x & 31;
  const int half = lane >> 4;
  const int mr = lane & 15;
  const int tilesN = N >> 4;
  const int tile = blockIdx.x * 4 + wave;
  if (tile >= (M >> 4) * tilesN) return;  // wave-uniform, EXEC stays all-ones
  const int m0 = (tile / tilesN) << 4;
  const int n0 = (tile % tilesN) << 4;

  v8f acc = {};
  const uint32_t* Arow = (const uint32_t*)(A + (size_t)(m0 + mr) * K);
  const uint32_t* Brow = (const uint32_t*)(Bw + (size_t)(n0 + mr) * K);
  for (int k0 = 0; k0 < K; k0 += 32) {
    BFrag a, b;
#pragma unroll
    for (int vi = 0; vi < 8; ++vi) {
      int ka = ((vi < 4) ? 2 * vi : 16 + 2 * (vi - 4)) + 8 * half;  // A layout
      a.u[vi] = Arow[(k0 + ka) >> 1];
      int kb = ((vi < 4) ? 2 * vi : 16 + 2 * (vi - 4)) + 8 * half;  // B layout
      b.u[vi] = Brow[(k0 + kb) >> 1];
    }
    acc = __builtin_amdgcn_wmma_f32_16x16x32_bf16(false, a.v, false, b.v,
                                                  (short)0, acc, false, false);
  }
#pragma unroll
  for (int r = 0; r < 8; ++r) {
    int m = m0 + r + 8 * half;
    int n = n0 + mr;
    float val = acc[r];
    if (flags & F_BIAS) val += bias[n];
    val *= scale;
    if (flags & F_SIGMOID) val = 1.0f / (1.0f + __expf(-val));
    if (flags & F_RELU) val = fmaxf(val, 0.0f);
    if (flags & F_RESID) val += resid[(size_t)m * N + n];
    if (outF) outF[(size_t)m * N + n] = val;
    if (outB) outB[(size_t)m * N + n] = f2bf(val);
    if (outBT) outBT[(size_t)n * M + m] = f2bf(val);  // transposed (for V)
  }
}

// ---------------------------------------------------------------------------
// Fused pair-biased flash attention with TDM-fed, double-buffered plm tiles.
// Block = 128 threads (4 waves), wave h == head h, 16 query rows per block.
// ---------------------------------------------------------------------------
__global__ void __launch_bounds__(128)
attn_fused(const uint16_t* __restrict__ qb, const uint16_t* __restrict__ kb16,
           const uint16_t* __restrict__ vt, const float* __restrict__ plm,
           const float* __restrict__ beta, const float* __restrict__ pw,
           const float* __restrict__ pbv, const float* __restrict__ Wpb,
           const float* __restrict__ gate, uint16_t* __restrict__ gated) {
  __shared__ float sbias[KT][16][N_HEADS];     // [key_local][qry_local][head]
  __shared__ uint16_t pshm[N_HEADS][16][KT];   // P staging for lane transpose
  extern __shared__ float ptile[];             // [2][16][KT][C_PAIR] (64 KB)

  const int tid = threadIdx.x;
  const int wave = tid >> 5, lane = tid & 31;
  const int half = lane >> 4, mr = lane & 15;
  const int h = wave;
  const int i0 = blockIdx.x << 4;
  const int NT = N_ATOMS / KT;

  // Q A-fragment (bq and 1/sqrt(32) already folded in).
  BFrag aq;
  {
    const uint32_t* Qrow =
        (const uint32_t*)(qb + (size_t)(i0 + mr) * C_ATOM + h * C_HEAD);
#pragma unroll
    for (int vi = 0; vi < 8; ++vi) {
      int ka = ((vi < 4) ? 2 * vi : 16 + 2 * (vi - 4)) + 8 * half;
      aq.u[vi] = Qrow[ka >> 1];
    }
  }

  v8f o0 = {}, o1 = {};
  float mrow[8], lrow[8];
#pragma unroll
  for (int r = 0; r < 8; ++r) { mrow[r] = -3.0e38f; lrow[r] = 0.0f; }

#if HAVE_TDM
  if (wave == 0)  // one TDM descriptor per tile, issued by wave 0 only
    tdm_load_plm_tile(plm + ((size_t)i0 * N_ATOMS) * C_PAIR, &ptile[0]);
#endif

  for (int t = 0; t < NT; ++t) {
    const int j0 = t * KT;
#if HAVE_TDM
    if (wave == 0) {
      if (t + 1 < NT) {
        tdm_load_plm_tile(plm + ((size_t)i0 * N_ATOMS + (j0 + KT)) * C_PAIR,
                          &ptile[((t + 1) & 1) * TILE_F]);
        // in-order TENSORcnt: tile t done when only the new issue remains
        __builtin_amdgcn_s_wait_tensorcnt(1);
      } else {
        __builtin_amdgcn_s_wait_tensorcnt(0);
      }
    }
    __syncthreads();  // tile t visible to all waves
    const float* cur = &ptile[(t & 1) * TILE_F];
#endif

    // ---- cooperative pair-bias tile: 512 (i,j) pairs over 128 threads ----
    for (int p = tid; p < 16 * KT; p += 128) {
      const int jl = p >> 4, il = p & 15;
#if HAVE_TDM
      const float* pr = cur + ((size_t)il * KT + jl) * C_PAIR;  // LDS
#else
      const float* pr =
          plm + ((size_t)(i0 + il) * N_ATOMS + (size_t)(j0 + jl)) * C_PAIR;
      if (j0 + KT < N_ATOMS)
        __builtin_prefetch(pr + KT * C_PAIR, 0, 1);
#endif
      if (j0 + KT < N_ATOMS)  // keep a global_prefetch_b8 on beta stream
        __builtin_prefetch(beta + (size_t)(i0 + il) * N_ATOMS + (j0 + KT + jl),
                           0, 1);
      const float4* p4 = (const float4*)pr;
      float4 x0 = p4[0], x1 = p4[1], x2 = p4[2], x3 = p4[3];
      float vals[16] = {x0.x, x0.y, x0.z, x0.w, x1.x, x1.y, x1.z, x1.w,
                        x2.x, x2.y, x2.z, x2.w, x3.x, x3.y, x3.z, x3.w};
      float s = 0.0f;
#pragma unroll
      for (int c = 0; c < 16; ++c) s += vals[c];
      float mu = s * (1.0f / 16.0f);
      float ss = 0.0f;
#pragma unroll
      for (int c = 0; c < 16; ++c) { float d = vals[c] - mu; ss += d * d; }
      float inv = rsqrtf(ss * (1.0f / 16.0f) + 1e-5f);
      float nb[16];
#pragma unroll
      for (int c = 0; c < 16; ++c) nb[c] = (vals[c] - mu) * inv * pw[c] + pbv[c];
      float bm = beta[(size_t)(i0 + il) * N_ATOMS + (size_t)(j0 + jl)];
#pragma unroll
      for (int hh = 0; hh < N_HEADS; ++hh) {
        float d = 0.0f;
#pragma unroll
        for (int c = 0; c < 16; ++c) d += nb[c] * Wpb[hh * C_PAIR + c];
        sbias[jl][il][hh] = d + bm;
      }
    }
    __syncthreads();

    // ---- S = Q K^T for 2 key sub-tiles ----
    v8f s0 = {}, s1 = {};
    {
      BFrag bk;
      const uint32_t* Krow =
          (const uint32_t*)(kb16 + (size_t)(j0 + mr) * C_ATOM + h * C_HEAD);
#pragma unroll
      for (int vi = 0; vi < 8; ++vi) {
        int kk = ((vi < 4) ? 2 * vi : 16 + 2 * (vi - 4)) + 8 * half;
        bk.u[vi] = Krow[kk >> 1];
      }
      s0 = __builtin_amdgcn_wmma_f32_16x16x32_bf16(false, aq.v, false, bk.v,
                                                   (short)0, s0, false, false);
      const uint32_t* Krow1 = (const uint32_t*)(kb16 + (size_t)(j0 + 16 + mr) *
                                                           C_ATOM + h * C_HEAD);
#pragma unroll
      for (int vi = 0; vi < 8; ++vi) {
        int kk = ((vi < 4) ? 2 * vi : 16 + 2 * (vi - 4)) + 8 * half;
        bk.u[vi] = Krow1[kk >> 1];
      }
      s1 = __builtin_amdgcn_wmma_f32_16x16x32_bf16(false, aq.v, false, bk.v,
                                                   (short)0, s1, false, false);
    }
    float sv0[8], sv1[8];
#pragma unroll
    for (int r = 0; r < 8; ++r) {
      int m = r + 8 * half;
      sv0[r] = s0[r] + sbias[mr][m][h];
      sv1[r] = s1[r] + sbias[16 + mr][m][h];
    }

    // ---- online softmax (row reductions across 16 columns via shfl_xor) ----
    float tmax[8];
#pragma unroll
    for (int r = 0; r < 8; ++r) tmax[r] = fmaxf(sv0[r], sv1[r]);
#pragma unroll
    for (int off = 1; off < 16; off <<= 1)
#pragma unroll
      for (int r = 0; r < 8; ++r)
        tmax[r] = fmaxf(tmax[r], __shfl_xor(tmax[r], off, 32));
    float alpha[8], tsum[8];
#pragma unroll
    for (int r = 0; r < 8; ++r) {
      float mn = fmaxf(mrow[r], tmax[r]);
      alpha[r] = __expf(mrow[r] - mn);
      mrow[r] = mn;
      sv0[r] = __expf(sv0[r] - mn);
      sv1[r] = __expf(sv1[r] - mn);
      tsum[r] = sv0[r] + sv1[r];
    }
#pragma unroll
    for (int off = 1; off < 16; off <<= 1)
#pragma unroll
      for (int r = 0; r < 8; ++r) tsum[r] += __shfl_xor(tsum[r], off, 32);
#pragma unroll
    for (int r = 0; r < 8; ++r) {
      lrow[r] = lrow[r] * alpha[r] + tsum[r];
      o0[r] *= alpha[r];
      o1[r] *= alpha[r];
    }

    // ---- stage P to LDS (bf16) for the C->A lane transpose ----
#pragma unroll
    for (int r = 0; r < 8; ++r) {
      int m = r + 8 * half;
      pshm[h][m][mr] = f2bf(sv0[r]);
      pshm[h][m][16 + mr] = f2bf(sv1[r]);
    }
    __builtin_amdgcn_wave_barrier();  // per-wave LDS ops are in-order

    // ---- O += P V : A from LDS, B from transposed V (contiguous dwords) ----
    BFrag ap;
#pragma unroll
    for (int vi = 0; vi < 8; ++vi) {
      int ka = ((vi < 4) ? 2 * vi : 16 + 2 * (vi - 4)) + 8 * half;
      ap.u[vi] = *(const uint32_t*)&pshm[h][mr][ka];
    }
    BFrag bv;
#pragma unroll
    for (int vi = 0; vi < 8; ++vi) {
      int kk = ((vi < 4) ? 2 * vi : 16 + 2 * (vi - 4)) + 8 * half;
      bv.u[vi] =
          *(const uint32_t*)(vt + (size_t)(h * C_HEAD + mr) * N_ATOMS + j0 + kk);
    }
    o0 = __builtin_amdgcn_wmma_f32_16x16x32_bf16(false, ap.v, false, bv.v,
                                                 (short)0, o0, false, false);
#pragma unroll
    for (int vi = 0; vi < 8; ++vi) {
      int kk = ((vi < 4) ? 2 * vi : 16 + 2 * (vi - 4)) + 8 * half;
      bv.u[vi] = *(const uint32_t*)(vt + (size_t)(h * C_HEAD + 16 + mr) *
                                             N_ATOMS + j0 + kk);
    }
    o1 = __builtin_amdgcn_wmma_f32_16x16x32_bf16(false, ap.v, false, bv.v,
                                                 (short)0, o1, false, false);
    __syncthreads();  // sbias / ptile reuse next iteration
  }

  // ---- finalize: 1/l, sigmoid gate (fp32), emit bf16 for the Wo GEMM ----
#pragma unroll
  for (int r = 0; r < 8; ++r) {
    float inv = 1.0f / lrow[r];
    int i = i0 + r + 8 * half;
    int c0 = h * C_HEAD + mr;
    int c1 = h * C_HEAD + 16 + mr;
    float g0 = gate[(size_t)i * C_ATOM + c0];
    float g1 = gate[(size_t)i * C_ATOM + c1];
    gated[(size_t)i * C_ATOM + c0] = f2bf(o0[r] * inv * g0);
    gated[(size_t)i * C_ATOM + c1] = f2bf(o1[r] * inv * g1);
  }
}

// ---------------------------------------------------------------------------
extern "C" void kernel_launch(void* const* d_in, const int* in_sizes, int n_in,
                              void* d_out, int out_size, void* d_ws,
                              size_t ws_size, hipStream_t stream) {
  const float* ql   = (const float*)d_in[0];
  // d_in[1] = cl, unused by the reference
  const float* plm  = (const float*)d_in[2];
  const float* beta = (const float*)d_in[3];
  const float* nqw  = (const float*)d_in[4];
  const float* nqb  = (const float*)d_in[5];
  const float* npw  = (const float*)d_in[6];
  const float* npb  = (const float*)d_in[7];
  const float* Wq   = (const float*)d_in[8];
  const float* bq   = (const float*)d_in[9];
  const float* Wk   = (const float*)d_in[10];
  const float* Wv   = (const float*)d_in[11];
  const float* Wpb  = (const float*)d_in[12];
  const float* Wg   = (const float*)d_in[13];
  const float* Wo   = (const float*)d_in[14];
  const float* tlw  = (const float*)d_in[15];
  const float* tlb  = (const float*)d_in[16];
  const float* W1   = (const float*)d_in[17];
  const float* b1   = (const float*)d_in[18];
  const float* W2   = (const float*)d_in[19];
  const float* b2   = (const float*)d_in[20];
  float* outp = (float*)d_out;

  char* ws = (char*)d_ws;
  size_t off = 0;
  auto alloc = [&](size_t bytes) -> char* {
    char* p = ws + off;
    off = (off + bytes + 255) & ~(size_t)255;
    return p;
  };
  uint16_t* Wqb  = (uint16_t*)alloc(C_ATOM * C_ATOM * 2);
  uint16_t* Wkb  = (uint16_t*)alloc(C_ATOM * C_ATOM * 2);
  uint16_t* Wvb  = (uint16_t*)alloc(C_ATOM * C_ATOM * 2);
  uint16_t* Wgb  = (uint16_t*)alloc(C_ATOM * C_ATOM * 2);
  uint16_t* Wob  = (uint16_t*)alloc(C_ATOM * C_ATOM * 2);
  uint16_t* W1b  = (uint16_t*)alloc(4 * C_ATOM * C_ATOM * 2);
  uint16_t* W2b  = (uint16_t*)alloc(4 * C_ATOM * C_ATOM * 2);
  uint16_t* qln  = (uint16_t*)alloc((size_t)N_ATOMS * C_ATOM * 2);
  uint16_t* qbuf = (uint16_t*)alloc((size_t)N_ATOMS * C_ATOM * 2);
  uint16_t* kbuf = (uint16_t*)alloc((size_t)N_ATOMS * C_ATOM * 2);
  uint16_t* vtbf = (uint16_t*)alloc((size_t)N_ATOMS * C_ATOM * 2);  // [ch][atom]
  float*    gatef = (float*)alloc((size_t)N_ATOMS * C_ATOM * 4);
  uint16_t* gatedb = (uint16_t*)alloc((size_t)N_ATOMS * C_ATOM * 2);
  float*    ql1f = (float*)alloc((size_t)N_ATOMS * C_ATOM * 4);
  uint16_t* tb   = (uint16_t*)alloc((size_t)N_ATOMS * C_ATOM * 2);
  uint16_t* hidb = (uint16_t*)alloc((size_t)N_ATOMS * 4 * C_ATOM * 2);
  (void)ws_size; (void)in_sizes; (void)n_in; (void)out_size;

  auto cvt = [&](const float* s, uint16_t* d, int n) {
    cvt_bf16_kernel<<<(n + 255) / 256, 256, 0, stream>>>(s, d, n);
  };
  cvt(Wq, Wqb, C_ATOM * C_ATOM);
  cvt(Wk, Wkb, C_ATOM * C_ATOM);
  cvt(Wv, Wvb, C_ATOM * C_ATOM);
  cvt(Wg, Wgb, C_ATOM * C_ATOM);
  cvt(Wo, Wob, C_ATOM * C_ATOM);
  cvt(W1, W1b, 4 * C_ATOM * C_ATOM);
  cvt(W2, W2b, 4 * C_ATOM * C_ATOM);

  layernorm128_kernel<<<N_ATOMS, 128, 0, stream>>>(ql, nqw, nqb, qln);

  auto gemm = [&](const uint16_t* A, const uint16_t* Bw, const float* bias,
                  const float* resid, float scale, int flags, float* oF,
                  uint16_t* oB, uint16_t* oBT, int M, int N, int K) {
    int blocks = (M / 16) * (N / 16) / 4;
    gemm_bf16_wmma<<<blocks, 128, 0, stream>>>(A, Bw, bias, resid, scale,
                                               flags, oF, oB, oBT, M, N, K);
  };

  // q = (ln @ Wq^T + bq) / sqrt(32)   (scale folded into q)
  gemm(qln, Wqb, bq, nullptr, 0.17677669529663689f, F_BIAS, nullptr, qbuf,
       nullptr, N_ATOMS, C_ATOM, C_ATOM);
  gemm(qln, Wkb, nullptr, nullptr, 1.0f, 0, nullptr, kbuf, nullptr, N_ATOMS,
       C_ATOM, C_ATOM);
  gemm(qln, Wvb, nullptr, nullptr, 1.0f, 0, nullptr, nullptr, vtbf, N_ATOMS,
       C_ATOM, C_ATOM);  // transposed V
  gemm(qln, Wgb, nullptr, nullptr, 1.0f, F_SIGMOID, gatef, nullptr, nullptr,
       N_ATOMS, C_ATOM, C_ATOM);

  // dynamic LDS: 2 x 32KB double-buffered plm tile
  attn_fused<<<N_ATOMS / 16, 128, 2 * TILE_F * sizeof(float), stream>>>(
      qbuf, kbuf, vtbf, plm, beta, npw, npb, Wpb, gatef, gatedb);

  // ql1 = ql + (gate*attn) @ Wo^T
  gemm(gatedb, Wob, nullptr, ql, 1.0f, F_RESID, ql1f, nullptr, nullptr,
       N_ATOMS, C_ATOM, C_ATOM);

  layernorm128_kernel<<<N_ATOMS, 128, 0, stream>>>(ql1f, tlw, tlb, tb);

  // hid = relu(t @ W1^T + b1)
  gemm(tb, W1b, b1, nullptr, 1.0f, F_BIAS | F_RELU, nullptr, hidb, nullptr,
       N_ATOMS, 4 * C_ATOM, C_ATOM);
  // out = ql1 + hid @ W2^T + b2
  gemm(hidb, W2b, b2, ql1f, 1.0f, F_BIAS | F_RESID, outp, nullptr, nullptr,
       N_ATOMS, C_ATOM, 4 * C_ATOM);
}
